// CascadeClassifier_39084202393967
// MI455X (gfx1250) — matrix operand order repaired and checked by reading, not compiled, not run on hardware
//
#include <hip/hip_runtime.h>
#include <hip/hip_bf16.h>

typedef __attribute__((ext_vector_type(8)))  float  v8f;
typedef __attribute__((ext_vector_type(16))) __bf16 v16bf;

#define NNODES 50000
#define NEDGES 400000
#define NGRAPH 64
#define EPSBN  1e-5f

__device__ __forceinline__ unsigned int f32_to_bf16_rne(float f) {
    union { float f; unsigned int u; } x; x.f = f;
    unsigned int r = x.u + 0x7FFFu + ((x.u >> 16) & 1u);
    return r >> 16;
}

__device__ __forceinline__ float atomicMaxFloat(float* addr, float value) {
    int* ai = (int*)addr;
    int old = *ai;
    while (true) {
        float f = __int_as_float(old);
        if (f >= value) break;
        int assumed = old;
        old = atomicCAS(ai, assumed, __float_as_int(value));
        if (old == assumed) break;
    }
    return __int_as_float(old);
}

// ---------------------------------------------------------------------------
// Pack W (K x Cout, f32 row-major) into bf16 WMMA B-fragment layout:
//   idx = ((kt * nN + nt) * 32 + lane) * 16 + j
//   holds W[kt*32 + (j<8 ? 8*(lane>>4)+j : 16+8*(lane>>4)+j-8)][nt*16 + (lane&15)]
// One lane's fragment = 16 contiguous bf16 = 32 bytes.
// ---------------------------------------------------------------------------
__global__ void pack_w_bf16(const float* __restrict__ W,
                            unsigned short* __restrict__ Wp, int K, int Cout) {
    int idx = blockIdx.x * 256 + threadIdx.x;
    if (idx >= K * Cout) return;
    int j    = idx & 15;
    int l    = (idx >> 4) & 31;
    int tile = idx >> 9;
    int nN   = Cout >> 4;
    int nt   = tile % nN;
    int kt   = tile / nN;
    int k    = kt * 32 + (j < 8 ? (l >> 4) * 8 + j : 16 + (l >> 4) * 8 + (j - 8));
    int c    = nt * 16 + (l & 15);
    Wp[idx]  = (unsigned short)f32_to_bf16_rne(W[(size_t)k * Cout + c]);
}

// ---------------------------------------------------------------------------
// WMMA GEMM:  Out[NNODES x Cout] = A[NNODES x K] @ W[K x Cout]
// block = 256 thr = 8 waves; one 16-row tile per block; wave w covers
// columns [w*NT*16, (w+1)*NT*16). A staged in LDS pre-packed to bf16
// fragment order (double buffered); B from pre-packed bf16 weights.
// ---------------------------------------------------------------------------
template <int NT>
__global__ __launch_bounds__(256) void gemm_bf16_wmma(
    const float* __restrict__ A, const unsigned short* __restrict__ Wp,
    float* __restrict__ Out, int K, int Cout) {
    __shared__ __align__(32) unsigned int AsU[2][256];  // 2 x (512 bf16)

    const int tid    = threadIdx.x;
    const int wave   = tid >> 5;
    const int lane   = tid & 31;
    const int m0     = blockIdx.x * 16;
    const int lhalf  = lane >> 4;
    const int lmod   = lane & 15;
    const int nN     = Cout >> 4;
    const int ntBase = wave * NT;
    const int nK     = K >> 5;

    // Iteration-invariant staging coords: thread t fills packed elements
    // (2t, 2t+1) of the A tile -> lane sl, elems (sj, sj+1), which map to two
    // consecutive K values of row (sl&15).
    {
        // nothing
    }
    const int e     = tid << 1;
    const int sl    = e >> 4;
    const int sj    = e & 15;          // even
    const int srow  = sl & 15;
    const int sklo  = (sl >> 4) * 8;
    const int sv    = sj >> 1;
    const int sbase = (sv < 4) ? (sklo + 2 * sv) : (16 + sklo + 2 * (sv - 4));
    const float* aRow = A + (size_t)(m0 + srow) * K + sbase;

    v8f zero = {0.f, 0.f, 0.f, 0.f, 0.f, 0.f, 0.f, 0.f};
    v8f acc[NT];
#pragma unroll
    for (int nt = 0; nt < NT; ++nt) acc[nt] = zero;

    // stage tile 0
    {
        float2 f = *(const float2*)aRow;
        AsU[0][tid] = f32_to_bf16_rne(f.x) | (f32_to_bf16_rne(f.y) << 16);
    }

    int cur = 0;
    for (int kt = 0; kt < nK; ++kt) {
        __syncthreads();
        v16bf aF = *(const v16bf*)&AsU[cur][lane << 3];   // 32B aligned

        if (kt + 1 < nK) {  // stage next tile into the other buffer
            float2 f = *(const float2*)(aRow + (kt + 1) * 32);
            AsU[cur ^ 1][tid] = f32_to_bf16_rne(f.x) | (f32_to_bf16_rne(f.y) << 16);
        }

        const unsigned short* bp =
            Wp + (((size_t)kt * nN + ntBase) << 9) + (lane << 4);
#pragma unroll
        for (int nt = 0; nt < NT; ++nt) {
            v16bf bF = *(const v16bf*)(bp + ((size_t)nt << 9));
            acc[nt] = __builtin_amdgcn_wmma_f32_16x16x32_bf16(
                false, aF, false, bF, (short)0, acc[nt], false, false);
        }
        cur ^= 1;
    }

    // D layout: VGPR v -> row v + 8*(lane>>4); col = lane&15
    const int cBase = (ntBase << 4) + lmod;
#pragma unroll
    for (int nt = 0; nt < NT; ++nt) {
        float* orow = Out + (size_t)(m0 + 8 * lhalf) * Cout + cBase + nt * 16;
#pragma unroll
        for (int v = 0; v < 8; ++v)
            orow[(size_t)v * Cout] = acc[nt][v];
    }
}

// ---------------------------------------------------------------------------
// Utility / graph kernels
// ---------------------------------------------------------------------------
__global__ void fill_f32(float* __restrict__ p, float v, size_t n) {
    size_t i = (size_t)blockIdx.x * blockDim.x + threadIdx.x;
    if (i < n) p[i] = v;
}

__global__ void deg_accum(const int* __restrict__ dst, float* __restrict__ deg, int E) {
    int i = blockIdx.x * 256 + threadIdx.x;
    if (i < E) atomicAdd(&deg[dst[i]], 1.0f);
}

__global__ void deg_to_dis(float* __restrict__ deg, int n) {
    int i = blockIdx.x * 256 + threadIdx.x;
    if (i < n) deg[i] = rsqrtf(deg[i] + 1.0f);   // +1 self loop
}

// one block per (edge or self-loop); channels across threads
__global__ __launch_bounds__(256) void scatter_edges(
    const float* __restrict__ hw, const int* __restrict__ src, const int* __restrict__ dst,
    const float* __restrict__ dis, float* __restrict__ agg, int C, int E) {
    int e = blockIdx.x;
    int s, d; float w;
    if (e < E) { s = src[e]; d = dst[e]; w = dis[s] * dis[d]; }
    else       { s = d = e - E; float q = dis[s]; w = q * q; }
    const float* row = hw + (size_t)s * C;
    float* orow = agg + (size_t)d * C;
    for (int c = threadIdx.x; c < C; c += 256)
        atomicAdd(&orow[c], row[c] * w);
}

// one block per column: biased mean/var over NNODES rows -> scale/shift
__global__ __launch_bounds__(256) void bn_stats(
    const float* __restrict__ h, const float* __restrict__ gamma,
    const float* __restrict__ beta, float* __restrict__ scl,
    float* __restrict__ shf, int C) {
    __shared__ float s1[256], s2[256];
    int c = blockIdx.x, tid = threadIdx.x;
    float a = 0.f, b = 0.f;
    for (int r = tid; r < NNODES; r += 256) {
        float v = h[(size_t)r * C + c];
        a += v; b += v * v;
    }
    s1[tid] = a; s2[tid] = b;
    __syncthreads();
    for (int off = 128; off > 0; off >>= 1) {
        if (tid < off) { s1[tid] += s1[tid + off]; s2[tid] += s2[tid + off]; }
        __syncthreads();
    }
    if (tid == 0) {
        float inv = 1.0f / (float)NNODES;
        float m   = s1[0] * inv;
        float var = s2[0] * inv - m * m;
        float s   = gamma[c] * rsqrtf(var + EPSBN);
        scl[c] = s;
        shf[c] = beta[c] - m * s;
    }
}

__global__ void bn_apply(float* __restrict__ h, const float* __restrict__ scl,
                         const float* __restrict__ shf, int C, size_t total, int relu) {
    size_t i = (size_t)blockIdx.x * 256 + threadIdx.x;
    if (i >= total) return;
    int c = (int)(i % (size_t)C);
    float v = h[i] * scl[c] + shf[c];
    if (relu) v = v > 0.f ? v : 0.f;
    h[i] = v;
}

// one block per node, 128 channels
__global__ __launch_bounds__(128) void pool_kernel(
    const float* __restrict__ h, const int* __restrict__ batch,
    float* __restrict__ pmean, float* __restrict__ pmax, float* __restrict__ cnt) {
    int i = blockIdx.x;
    int b = batch[i];
    if (threadIdx.x == 0) atomicAdd(&cnt[b], 1.0f);
    int c = threadIdx.x;
    float v = h[(size_t)i * 128 + c];
    atomicAdd(&pmean[b * 128 + c], v);
    atomicMaxFloat(&pmax[b * 128 + c], v);
}

__global__ __launch_bounds__(128) void gbuild_kernel(
    const float* __restrict__ pmean, const float* __restrict__ pmax,
    const float* __restrict__ cnt, float* __restrict__ g) {
    int b = blockIdx.x, c = threadIdx.x;
    float n = cnt[b]; if (n < 1.f) n = 1.f;
    g[b * 256 + c]       = pmean[b * 128 + c] / n;
    g[b * 256 + 128 + c] = pmax[b * 128 + c];
}

// classifier: Linear(256->64) -> ReLU -> BN -> Linear(64->2), one block
__global__ __launch_bounds__(256) void classifier_kernel(
    const float* __restrict__ g, const float* __restrict__ Wc1,
    const float* __restrict__ bc1, const float* __restrict__ gamma,
    const float* __restrict__ beta, const float* __restrict__ Wc2,
    const float* __restrict__ bc2, float* __restrict__ out) {
    __shared__ float hc[64 * 64];
    __shared__ float sc[64], sh[64];
    int tid = threadIdx.x;
    for (int idx = tid; idx < 64 * 64; idx += 256) {
        int r = idx >> 6, c = idx & 63;
        float s = bc1[c];
        for (int k = 0; k < 256; ++k) s += g[r * 256 + k] * Wc1[k * 64 + c];
        hc[idx] = s > 0.f ? s : 0.f;
    }
    __syncthreads();
    if (tid < 64) {
        float s = 0.f, q = 0.f;
        for (int r = 0; r < 64; ++r) { float v = hc[r * 64 + tid]; s += v; q += v * v; }
        float m   = s * (1.f / 64.f);
        float var = q * (1.f / 64.f) - m * m;
        float scl = gamma[tid] * rsqrtf(var + EPSBN);
        sc[tid] = scl;
        sh[tid] = beta[tid] - m * scl;
    }
    __syncthreads();
    if (tid < 128) {
        int r = tid >> 1, c = tid & 1;
        float s = bc2[c];
        for (int k = 0; k < 64; ++k)
            s += (hc[r * 64 + k] * sc[k] + sh[k]) * Wc2[k * 2 + c];
        out[r * 2 + c] = s;
    }
}

// ---------------------------------------------------------------------------
extern "C" void kernel_launch(void* const* d_in, const int* in_sizes, int n_in,
                              void* d_out, int out_size, void* d_ws, size_t ws_size,
                              hipStream_t stream) {
    (void)in_sizes; (void)n_in; (void)out_size; (void)ws_size;

    const float* x     = (const float*)d_in[0];
    const int*   eidx  = (const int*)d_in[1];
    const int*   batch = (const int*)d_in[2];
    const float* W0  = (const float*)d_in[3];
    const float* g0  = (const float*)d_in[5];
    const float* be0 = (const float*)d_in[6];
    const float* W1  = (const float*)d_in[7];
    const float* g1  = (const float*)d_in[9];
    const float* be1 = (const float*)d_in[10];
    const float* W2  = (const float*)d_in[11];
    const float* g2  = (const float*)d_in[13];
    const float* be2 = (const float*)d_in[14];
    const float* Wc1 = (const float*)d_in[15];
    const float* bc1 = (const float*)d_in[16];
    const float* gc  = (const float*)d_in[17];
    const float* bec = (const float*)d_in[18];
    const float* Wc2 = (const float*)d_in[19];
    const float* bc2 = (const float*)d_in[20];
    float* out = (float*)d_out;

    const int* src = eidx;
    const int* dst = eidx + NEDGES;

    char* ws = (char*)d_ws;
    const size_t BUF = (size_t)NNODES * 512 * sizeof(float); // 102.4 MB
    float* bufA  = (float*)(ws);                 // h / aggregation buffer
    float* bufB  = (float*)(ws + BUF);           // hw = h @ W
    float* dis   = (float*)(ws + 2 * BUF);       // deg -> deg^-1/2   (N floats)
    char*  p     = ws + 2 * BUF + ((size_t)NNODES * 4 + 255) / 256 * 256;
    float* scl   = (float*)(p);        p += 512 * 4;
    float* shf   = (float*)(p);        p += 512 * 4;
    float* pmean = (float*)(p);        p += 64 * 128 * 4;
    float* pmax  = (float*)(p);        p += 64 * 128 * 4;
    float* cnt   = (float*)(p);        p += 256;
    float* gfeat = (float*)(p);        p += 64 * 256 * 4;
    unsigned short* Wp = (unsigned short*)(p);   // packed bf16 weights (<=768KB)

    const int MT = NNODES / 16;          // 3125 row tiles
    const int EN = NEDGES + NNODES;      // edges + self loops

    // degrees -> deg^-1/2
    fill_f32<<<(NNODES + 255) / 256, 256, 0, stream>>>(dis, 0.0f, (size_t)NNODES);
    deg_accum<<<(NEDGES + 255) / 256, 256, 0, stream>>>(dst, dis, NEDGES);
    deg_to_dis<<<(NNODES + 255) / 256, 256, 0, stream>>>(dis, NNODES);

    struct Layer { const float* W; const float* gam; const float* bet; int K; int C; int NT; int relu; };
    Layer layers[3] = {
        {W0, g0, be0, 768, 512, 4, 1},
        {W1, g1, be1, 512, 256, 2, 1},
        {W2, g2, be2, 256, 128, 1, 0},
    };

    const float* hin = x;
    for (int li = 0; li < 3; ++li) {
        Layer& L = layers[li];
        int wTot = L.K * L.C;
        pack_w_bf16<<<(wTot + 255) / 256, 256, 0, stream>>>(L.W, Wp, L.K, L.C);

        if (L.NT == 4)
            gemm_bf16_wmma<4><<<MT, 256, 0, stream>>>(hin, Wp, bufB, L.K, L.C);
        else if (L.NT == 2)
            gemm_bf16_wmma<2><<<MT, 256, 0, stream>>>(hin, Wp, bufB, L.K, L.C);
        else
            gemm_bf16_wmma<1><<<MT, 256, 0, stream>>>(hin, Wp, bufB, L.K, L.C);

        size_t total = (size_t)NNODES * L.C;
        fill_f32<<<(unsigned)((total + 255) / 256), 256, 0, stream>>>(bufA, 0.0f, total);
        scatter_edges<<<EN, 256, 0, stream>>>(bufB, src, dst, dis, bufA, L.C, NEDGES);
        bn_stats<<<L.C, 256, 0, stream>>>(bufA, L.gam, L.bet, scl, shf, L.C);
        bn_apply<<<(unsigned)((total + 255) / 256), 256, 0, stream>>>(bufA, scl, shf, L.C, total, L.relu);
        hin = bufA;
    }

    // pooling
    fill_f32<<<(64 * 128 + 255) / 256, 256, 0, stream>>>(pmean, 0.0f, (size_t)64 * 128);
    fill_f32<<<(64 * 128 + 255) / 256, 256, 0, stream>>>(pmax, -__builtin_huge_valf(), (size_t)64 * 128);
    fill_f32<<<1, 64, 0, stream>>>(cnt, 0.0f, (size_t)64);
    pool_kernel<<<NNODES, 128, 0, stream>>>(bufA, batch, pmean, pmax, cnt);
    gbuild_kernel<<<NGRAPH, 128, 0, stream>>>(pmean, pmax, cnt, gfeat);

    // classifier
    classifier_kernel<<<1, 256, 0, stream>>>(gfeat, Wc1, bc1, gc, bec, Wc2, bc2, out);
}
